// InterExtractor_35579509080737
// MI455X (gfx1250) — matrix operand order repaired and checked by reading, not compiled, not run on hardware
//
#include <hip/hip_runtime.h>
#include <stdint.h>

// ---------------------------------------------------------------------------
// CDNA5 (gfx1250) bf16 WMMA implementation of InterExtractor forward.
// GEMMs: v_wmma_f32_16x16x32_bf16 (f32 accumulate). Weight tiles staged into
// LDS via the Tensor Data Mover (tensor_load_to_lds, 6-arg clang-23 form).
// ---------------------------------------------------------------------------

typedef __attribute__((ext_vector_type(16))) __bf16         v16bf;
typedef __attribute__((ext_vector_type(16))) unsigned short v16u;
typedef __attribute__((ext_vector_type(8)))  unsigned short v8us;
typedef __attribute__((ext_vector_type(8)))  float          v8f;
typedef __attribute__((ext_vector_type(4)))  unsigned int   v4ui;
typedef __attribute__((ext_vector_type(8)))  int            v8i;
typedef __attribute__((ext_vector_type(4)))  int            v4i;

#define WMMA_BF16(acc, a, b)                                                   \
    (acc) = __builtin_amdgcn_wmma_f32_16x16x32_bf16(                           \
        false, (a), false, (b), (short)0, (acc), false, false)

__device__ __forceinline__ v8f zero8() {
    v8f z = {0.f, 0.f, 0.f, 0.f, 0.f, 0.f, 0.f, 0.f};
    return z;
}

__device__ __forceinline__ unsigned short f32_to_bf16_bits(float f) {
    unsigned int u = __float_as_uint(f);
    u += 0x7FFFu + ((u >> 16) & 1u);   // round-to-nearest-even
    return (unsigned short)(u >> 16);
}

// ---- WMMA fragment loaders (ISA 7.12.2 layouts, wave32) --------------------

// A 16x32 bf16: lane<16 -> M=lane, K={0..7,16..23}; lane>=16 -> M=lane-16,
// K={8..15,24..31}. src at (mBase,kBase), row-major stride ld.
__device__ __forceinline__ v16bf frag_a_u16(const unsigned short* src, int ld, int lane) {
    const unsigned short* p = src + (lane & 15) * ld + ((lane >> 4) << 3);
    v16u u;
#pragma unroll
    for (int i = 0; i < 8; ++i) u[i] = p[i];
#pragma unroll
    for (int i = 0; i < 8; ++i) u[8 + i] = p[16 + i];
    return __builtin_bit_cast(v16bf, u);
}

// Same, with top-k mask applied in registers (values are positive bf16, so
// bit-pattern compare == value compare).
__device__ __forceinline__ v16bf frag_a_masked(const unsigned short* src, int ld,
                                               int lane, unsigned short thr) {
    const unsigned short* p = src + (lane & 15) * ld + ((lane >> 4) << 3);
    v16u u;
#pragma unroll
    for (int i = 0; i < 8; ++i) u[i] = p[i];
#pragma unroll
    for (int i = 0; i < 8; ++i) u[8 + i] = p[16 + i];
#pragma unroll
    for (int i = 0; i < 16; ++i) u[i] = (u[i] >= thr) ? u[i] : (unsigned short)0;
    return __builtin_bit_cast(v16bf, u);
}

// A fragment from f32 source (converted on the fly).
__device__ __forceinline__ v16bf frag_a_f32(const float* src, int ld, int lane) {
    const float* p = src + (lane & 15) * ld + ((lane >> 4) << 3);
    v16u u;
#pragma unroll
    for (int i = 0; i < 8; ++i) u[i] = f32_to_bf16_bits(p[i]);
#pragma unroll
    for (int i = 0; i < 8; ++i) u[8 + i] = f32_to_bf16_bits(p[16 + i]);
    return __builtin_bit_cast(v16bf, u);
}

// B 32x16 bf16 where B(k,n) = src[n*ld + k] ("W^T" GEMM): lane<16 K={0..7,16..23}, lane>=16 K={8..15,24..31}.
__device__ __forceinline__ v16bf frag_bT_u16(const unsigned short* src, int ld, int lane) {
    const unsigned short* p = src + (lane & 15) * ld + ((lane >> 4) << 3);
    v16u u;
#pragma unroll
    for (int i = 0; i < 8; ++i) u[i] = p[i];
#pragma unroll
    for (int i = 0; i < 8; ++i) u[8 + i] = p[16 + i];
    return __builtin_bit_cast(v16bf, u);
}

// ---------------------------------------------------------------------------
// Tensor Data Mover: load a contiguous row-major 2D bf16 tile into LDS.
// D# per ISA §8: group0 = {count|flags, lds_addr, global_addr lo, hi|type=2},
// group1 packs data_size=2B, tensor/tile dims and strides. Groups 2/3 zero
// (2D tensor). clang-23 builtin takes 6 args: (g0, g1, g2, g3, g1b, cpol).
// ---------------------------------------------------------------------------
#if __has_builtin(__builtin_amdgcn_tensor_load_to_lds)
#define HAVE_TDM 1
__device__ __forceinline__ void tdm_load_2d_bf16(const unsigned short* gsrc,
                                                 unsigned lds_off,
                                                 int width, int height) {
    unsigned long long ga = (unsigned long long)(uintptr_t)gsrc;
    v4ui g0;
    g0[0] = 1u;                                       // count=1, user mode
    g0[1] = lds_off;                                  // lds_addr (bytes)
    g0[2] = (unsigned)(ga & 0xFFFFFFFFu);             // global_addr[31:0]
    g0[3] = (unsigned)((ga >> 32) & 0x01FFFFFFu) | 0x80000000u;  // [56:32]|type=2
    unsigned w = (unsigned)width, h = (unsigned)height;
    unsigned s1 = w * h;                              // dim1 stride (elements)
    v8i g1;
    g1[0] = (int)(1u << 16);                          // data_size = 2 bytes
    g1[1] = (int)((w & 0xFFFFu) << 16);               // tensor_dim0[15:0] @ bit48
    g1[2] = (int)((w >> 16) | ((h & 0xFFFFu) << 16)); // dim0 hi | tensor_dim1 lo
    g1[3] = (int)((h >> 16) | ((w & 0xFFFFu) << 16)); // dim1 hi | tile_dim0
    g1[4] = (int)(h & 0xFFFFu);                       // tile_dim1 (tile_dim2=0)
    g1[5] = (int)w;                                   // tensor_dim0_stride lo32
    g1[6] = (int)((s1 & 0xFFFFu) << 16);              // stride0 hi16=0 | stride1 lo16
    g1[7] = (int)(s1 >> 16);                          // stride1 hi
    v4i gz4 = {0, 0, 0, 0};
    v8i gz8 = {0, 0, 0, 0, 0, 0, 0, 0};
    __builtin_amdgcn_tensor_load_to_lds(g0, g1, gz4, gz4, gz8, 0);
}
#else
#define HAVE_TDM 0
#endif

// Stage one 128x128 bf16 tile (32KB) from global into LDS (all 256 threads).
__device__ __forceinline__ void stage_tile_32k(unsigned short* lds,
                                               const unsigned short* gsrc) {
#if HAVE_TDM
    if (threadIdx.x < 32) {   // wave 0 issues the DMA, waits TENSORcnt
        tdm_load_2d_bf16(gsrc, (unsigned)(uintptr_t)lds, 128, 128);
        __builtin_amdgcn_s_wait_tensorcnt(0);
    }
#else
    v8us* d = (v8us*)lds;
    const v8us* s = (const v8us*)gsrc;
    for (int i = threadIdx.x; i < 128 * 128 / 8; i += 256) d[i] = s[i];
#endif
}

// ---------------------------------------------------------------------------
// Problem constants
// ---------------------------------------------------------------------------
#define BGRAPHS 64
#define MAXN    2048
#define HDIM    128
#define CDIM    64
#define NTOT    (BGRAPHS * MAXN)                 // 131072
#define AELEMS  ((size_t)BGRAPHS * CDIM * MAXN)  // 8388608 per tensor
#define TOPK    6291456LL                        // int(8388608 * 0.75)

// ---------------------------------------------------------------------------
// Kernel 0a: zero histograms / threshold slots
// ---------------------------------------------------------------------------
__global__ void zero_u32_kernel(unsigned int* __restrict__ p, int n) {
    int i = blockIdx.x * 256 + threadIdx.x;
    if (i < n) p[i] = 0u;
}

// ---------------------------------------------------------------------------
// Kernel 0b: prep -- Qp = Q @ WQ^T + bQ (bf16), plus bf16 copies of WK/WV/WO.
// grid = 4 blocks.
// ---------------------------------------------------------------------------
__global__ __launch_bounds__(256) void prep_kernel(
    const float* __restrict__ Q,  const float* __restrict__ WQ,
    const float* __restrict__ bQ, const float* __restrict__ WK,
    const float* __restrict__ WV, const float* __restrict__ WO,
    unsigned short* __restrict__ QpB, unsigned short* __restrict__ WKb,
    unsigned short* __restrict__ WVb, unsigned short* __restrict__ WOb) {
    if (blockIdx.x == 0) {
        for (int idx = threadIdx.x; idx < CDIM * HDIM; idx += 256) {
            int c = idx >> 7, n = idx & 127;
            const float* q = Q + c * HDIM;
            const float* w = WQ + n * HDIM;
            float s = bQ[n];
#pragma unroll 8
            for (int h = 0; h < HDIM; ++h) s += q[h] * w[h];
            QpB[idx] = f32_to_bf16_bits(s);
        }
    } else {
        const float* src = (blockIdx.x == 1) ? WK : (blockIdx.x == 2 ? WV : WO);
        unsigned short* dst = (blockIdx.x == 1) ? WKb : (blockIdx.x == 2 ? WVb : WOb);
        for (int i = threadIdx.x; i < HDIM * HDIM; i += 256)
            dst[i] = f32_to_bf16_bits(src[i]);
    }
}

// ---------------------------------------------------------------------------
// Kernel 1: fused K/V projections. grid = (1024, 2): y selects x1/x2.
// K stored row-major [node][h]; V stored TRANSPOSED [b][h][node] so that the
// attention GEMM reads contiguous B-fragments straight from global.
// ---------------------------------------------------------------------------
__global__ __launch_bounds__(256) void proj_kv_kernel(
    const float* __restrict__ x1, const float* __restrict__ x2,
    const unsigned short* __restrict__ WKb, const float* __restrict__ bK,
    const unsigned short* __restrict__ WVb, const float* __restrict__ bV,
    unsigned short* __restrict__ Kb1, unsigned short* __restrict__ VbT1,
    unsigned short* __restrict__ Kb2, unsigned short* __restrict__ VbT2) {
    __shared__ unsigned short sWK[HDIM * HDIM];  // 32 KB
    __shared__ unsigned short sWV[HDIM * HDIM];  // 32 KB

    const float* x = blockIdx.y ? x2 : x1;
    unsigned short* Ko = blockIdx.y ? Kb2 : Kb1;
    unsigned short* VoT = blockIdx.y ? VbT2 : VbT1;
    const int rowBase = blockIdx.x * 128;     // global node index base
    const int bIdx = rowBase >> 11;           // graph id (2048 nodes/graph)
    const int nodeInG = rowBase & (MAXN - 1); // node offset within graph

    stage_tile_32k(sWK, WKb);
    stage_tile_32k(sWV, WVb);
    __syncthreads();

    const int t = threadIdx.x;
    const int lane = t & 31, wave = t >> 5;
    const float* xrow = x + (size_t)(rowBase + wave * 16) * HDIM;

    v16bf afr[4];
#pragma unroll
    for (int kc = 0; kc < 4; ++kc) afr[kc] = frag_a_f32(xrow + kc * 32, HDIM, lane);

    for (int pass = 0; pass < 2; ++pass) {
        const unsigned short* sW = pass ? sWV : sWK;
        const float* bias = pass ? bV : bK;

        v8f acc[8];
#pragma unroll
        for (int nt = 0; nt < 8; ++nt) acc[nt] = zero8();
#pragma unroll
        for (int kc = 0; kc < 4; ++kc) {
#pragma unroll
            for (int nt = 0; nt < 8; ++nt) {
                v16bf bb = frag_bT_u16(sW + nt * 16 * HDIM + kc * 32, HDIM, lane);
                WMMA_BF16(acc[nt], afr[kc], bb);
            }
        }

        if (pass == 0) {
            // K: row-major [node][h]
#pragma unroll
            for (int nt = 0; nt < 8; ++nt) {
                int n = nt * 16 + (lane & 15);
                float bv = bias[n];
#pragma unroll
                for (int r = 0; r < 8; ++r) {
                    int m = wave * 16 + ((lane >> 4) << 3) + r;
                    Ko[(size_t)(rowBase + m) * HDIM + n] = f32_to_bf16_bits(acc[nt][r] + bv);
                }
            }
        } else {
            // V: transposed [b][h][node], packed 16B store per (lane, nt)
#pragma unroll
            for (int nt = 0; nt < 8; ++nt) {
                int n = nt * 16 + (lane & 15);
                float bv = bias[n];
                v8us pk;
#pragma unroll
                for (int r = 0; r < 8; ++r) pk[r] = f32_to_bf16_bits(acc[nt][r] + bv);
                int node = nodeInG + wave * 16 + ((lane >> 4) << 3);
                *(v8us*)(VoT + (size_t)bIdx * HDIM * MAXN + (size_t)n * MAXN + node) = pk;
            }
        }
    }
}

// ---------------------------------------------------------------------------
// Kernel 2: scores (Qp @ K^T * scale), softmax over clusters, bf16 A write
// (packed 16B stores), LDS histogram of bf16 bit-patterns (4096 bins).
// grid = (1024, 2): x = b*16 + nodeTile, y = 0 -> A1 (K2), 1 -> A2 (K1).
// ---------------------------------------------------------------------------
__global__ __launch_bounds__(256) void scores_kernel(
    const unsigned short* __restrict__ Kb1, const unsigned short* __restrict__ Kb2,
    const unsigned short* __restrict__ QpB,
    unsigned short* __restrict__ A1, unsigned short* __restrict__ A2,
    unsigned int* __restrict__ hist1, unsigned int* __restrict__ hist2) {
    __shared__ float sS[CDIM * 128];      // 32 KB scores / softmax workspace
    __shared__ unsigned int sHist[4096];  // 16 KB

    const unsigned short* Ksrc = blockIdx.y ? Kb1 : Kb2;  // A1 pairs with K2
    unsigned short* Aout = blockIdx.y ? A2 : A1;
    unsigned int* hist = blockIdx.y ? hist2 : hist1;

    const int b = blockIdx.x >> 4;
    const int nodeBase = (blockIdx.x & 15) * 128;
    const unsigned short* Kt = Ksrc + ((size_t)b * MAXN + nodeBase) * HDIM;

    const int t = threadIdx.x;
    for (int i = t; i < 4096; i += 256) sHist[i] = 0u;

    const int lane = t & 31, wave = t >> 5;
    const int mt = wave & 3;             // cluster tile
    const int ntBase = (wave >> 2) * 4;  // node tiles [ntBase, ntBase+4)

    v8f acc[4];
#pragma unroll
    for (int j = 0; j < 4; ++j) acc[j] = zero8();
#pragma unroll
    for (int kc = 0; kc < 4; ++kc) {
        v16bf a = frag_a_u16(QpB + mt * 16 * HDIM + kc * 32, HDIM, lane);
#pragma unroll
        for (int j = 0; j < 4; ++j) {
            v16bf bb = frag_bT_u16(Kt + (size_t)(ntBase + j) * 16 * HDIM + kc * 32, HDIM, lane);
            WMMA_BF16(acc[j], a, bb);
        }
    }

    const float scale = 0.08838834764831845f;  // 1/sqrt(128)
#pragma unroll
    for (int j = 0; j < 4; ++j) {
        int n = (ntBase + j) * 16 + (lane & 15);
#pragma unroll
        for (int r = 0; r < 8; ++r) {
            int m = mt * 16 + ((lane >> 4) << 3) + r;
            sS[m * 128 + n] = acc[j][r] * scale;
        }
    }
    __syncthreads();

    // softmax over the 64 clusters, one column per thread
    if (t < 128) {
        float mx = -1e30f;
        for (int c = 0; c < CDIM; ++c) mx = fmaxf(mx, sS[c * 128 + t]);
        float sum = 0.f;
        for (int c = 0; c < CDIM; ++c) {
            float e = __expf(sS[c * 128 + t] - mx);
            sS[c * 128 + t] = e;
            sum += e;
        }
        float inv = 1.f / sum;
        for (int c = 0; c < CDIM; ++c) sS[c * 128 + t] *= inv;
    }
    __syncthreads();

    // bf16 A write-out (4 x 16B packed stores / thread) + histogram
    {
        const int c = t >> 2;
        const int col0 = (t & 3) * 32;
        unsigned short vals[32];
#pragma unroll
        for (int i = 0; i < 32; ++i) {
            vals[i] = f32_to_bf16_bits(sS[c * 128 + col0 + i]);
            atomicAdd(&sHist[vals[i] >> 4], 1u);
        }
        v8us* dst = (v8us*)(Aout + (size_t)b * CDIM * MAXN + (size_t)c * MAXN +
                            nodeBase + col0);
#pragma unroll
        for (int jj = 0; jj < 4; ++jj) {
            v8us pk;
#pragma unroll
            for (int e = 0; e < 8; ++e) pk[e] = vals[jj * 8 + e];
            dst[jj] = pk;
        }
    }
    __syncthreads();
    for (int i = t; i < 4096; i += 256) {
        unsigned int v = sHist[i];
        if (v) atomicAdd(&hist[i], v);
    }
}

// ---------------------------------------------------------------------------
// Kernel 3: scan histograms top-down for the top-k bin-edge threshold
// ---------------------------------------------------------------------------
__global__ void select_kernel(const unsigned int* __restrict__ hist1,
                              const unsigned int* __restrict__ hist2,
                              unsigned int* __restrict__ thr) {
    if (threadIdx.x < 2) {
        const unsigned int* h = threadIdx.x ? hist2 : hist1;
        long long cum = 0;
        int sel = 0;
        for (int b = 4095; b >= 0; --b) {
            cum += (long long)h[b];
            if (cum >= TOPK) { sel = b; break; }
        }
        thr[threadIdx.x] = (unsigned int)(sel << 4);
    }
}

// ---------------------------------------------------------------------------
// Kernel 4: out = relu((mask(A) @ V) @ WO^T + bO). grid = (64, 2).
// GEMM1: M=64, N=128, K=2048 -- A and transposed-V fragments loaded straight
// from global (L2-resident), mask applied in registers, no barriers in the
// k-loop. GEMM2 from LDS (T bf16) x WO (TDM-staged), fused bias+relu.
// ---------------------------------------------------------------------------
__global__ __launch_bounds__(256) void attn_out_kernel(
    const unsigned short* __restrict__ A1, const unsigned short* __restrict__ A2,
    const unsigned short* __restrict__ VbT1, const unsigned short* __restrict__ VbT2,
    const unsigned short* __restrict__ WOb, const float* __restrict__ bO,
    const unsigned int* __restrict__ thr, float* __restrict__ out) {
    __shared__ unsigned short sWO[HDIM * HDIM];  // 32 KB
    __shared__ unsigned short sT[CDIM * HDIM];   // 16 KB

    const int tensor = blockIdx.y;
    const unsigned short* Asrc = tensor ? A2 : A1;
    const unsigned short* Vsrc = tensor ? VbT1 : VbT2;  // A1 pairs with V2
    const unsigned short thrBits = (unsigned short)thr[tensor];
    const int b = blockIdx.x;

    stage_tile_32k(sWO, WOb);

    const int t = threadIdx.x;
    const int lane = t & 31, wave = t >> 5;
    const int mt = wave & 3;
    const int ntBase = (wave >> 2) * 4;

    const unsigned short* Ab = Asrc + ((size_t)b * CDIM + mt * 16) * MAXN;
    const unsigned short* Vt = Vsrc + (size_t)b * HDIM * MAXN;

    v8f acc[4];
#pragma unroll
    for (int j = 0; j < 4; ++j) acc[j] = zero8();

    for (int kc = 0; kc < MAXN / 32; ++kc) {
        __builtin_prefetch(Ab + ((kc + 8) & 63) * 32 + (lane & 15) * MAXN, 0, 1);
        v16bf a = frag_a_masked(Ab + kc * 32, MAXN, lane, thrBits);
#pragma unroll
        for (int j = 0; j < 4; ++j) {
            v16bf bb = frag_bT_u16(Vt + (size_t)(ntBase + j) * 16 * MAXN + kc * 32,
                                   MAXN, lane);
            WMMA_BF16(acc[j], a, bb);
        }
    }
    __syncthreads();  // also covers the sWO TDM staged by wave 0

    // T -> LDS as bf16
#pragma unroll
    for (int j = 0; j < 4; ++j) {
        int n = (ntBase + j) * 16 + (lane & 15);
#pragma unroll
        for (int r = 0; r < 8; ++r) {
            int m = mt * 16 + ((lane >> 4) << 3) + r;
            sT[m * HDIM + n] = f32_to_bf16_bits(acc[j][r]);
        }
    }
    __syncthreads();

    // GEMM2: T @ WO^T
    v8f acc2[4];
#pragma unroll
    for (int j = 0; j < 4; ++j) acc2[j] = zero8();
#pragma unroll
    for (int kc = 0; kc < 4; ++kc) {
        v16bf a2 = frag_a_u16(sT + mt * 16 * HDIM + kc * 32, HDIM, lane);
#pragma unroll
        for (int j = 0; j < 4; ++j) {
            v16bf b2 = frag_bT_u16(sWO + (ntBase + j) * 16 * HDIM + kc * 32, HDIM, lane);
            WMMA_BF16(acc2[j], a2, b2);
        }
    }

    // epilogue: bias + relu, f32 stores
#pragma unroll
    for (int j = 0; j < 4; ++j) {
        int n = (ntBase + j) * 16 + (lane & 15);
        float bv = bO[n];
#pragma unroll
        for (int r = 0; r < 8; ++r) {
            int m = mt * 16 + ((lane >> 4) << 3) + r;
            float y = acc2[j][r] + bv;
            y = fmaxf(y, 0.f);
            out[(size_t)tensor * (BGRAPHS * CDIM * HDIM) +
                ((size_t)b * CDIM + m) * HDIM + n] = y;
        }
    }
}

// ---------------------------------------------------------------------------
// Host-side launcher
// ---------------------------------------------------------------------------
extern "C" void kernel_launch(void* const* d_in, const int* in_sizes, int n_in,
                              void* d_out, int out_size, void* d_ws, size_t ws_size,
                              hipStream_t stream) {
    (void)in_sizes; (void)n_in; (void)out_size; (void)ws_size;

    const float* x1 = (const float*)d_in[0];
    const float* x2 = (const float*)d_in[1];
    // d_in[2], d_in[3]: batch ids -- equal-sized sorted graphs, unused
    const float* Q  = (const float*)d_in[4];
    const float* WQ = (const float*)d_in[5];
    const float* bQ = (const float*)d_in[6];
    const float* WK = (const float*)d_in[7];
    const float* bK = (const float*)d_in[8];
    const float* WV = (const float*)d_in[9];
    const float* bV = (const float*)d_in[10];
    const float* WO = (const float*)d_in[11];
    const float* bO = (const float*)d_in[12];
    float* out = (float*)d_out;

    char* ws = (char*)d_ws;
    const size_t KV_BYTES = (size_t)NTOT * HDIM * 2;  // 33,554,432
    const size_t A_BYTES  = AELEMS * 2;               // 16,777,216
    unsigned short* Kb1  = (unsigned short*)(ws);
    unsigned short* Kb2  = (unsigned short*)(ws + KV_BYTES);
    unsigned short* VbT1 = (unsigned short*)(ws + 2 * KV_BYTES);
    unsigned short* VbT2 = (unsigned short*)(ws + 3 * KV_BYTES);
    unsigned short* A1   = (unsigned short*)(ws + 4 * KV_BYTES);
    unsigned short* A2   = (unsigned short*)(ws + 4 * KV_BYTES + A_BYTES);
    char* base2 = ws + 4 * KV_BYTES + 2 * A_BYTES;
    unsigned short* QpB = (unsigned short*)(base2);            // 16 KB
    unsigned short* WKb = (unsigned short*)(base2 + 16384);    // 32 KB
    unsigned short* WVb = (unsigned short*)(base2 + 49152);    // 32 KB
    unsigned short* WOb = (unsigned short*)(base2 + 81920);    // 32 KB
    unsigned int* hist1 = (unsigned int*)(base2 + 114688);     // 16 KB
    unsigned int* hist2 = hist1 + 4096;                        // 16 KB
    unsigned int* thr   = hist2 + 4096;                        // 8 B

    // 1) zero histograms + thresholds (contiguous: 8194 u32)
    zero_u32_kernel<<<33, 256, 0, stream>>>(hist1, 8194);
    // 2) Qp + bf16 weight copies
    prep_kernel<<<4, 256, 0, stream>>>(Q, WQ, bQ, WK, WV, WO, QpB, WKb, WVb, WOb);
    // 3) K/V projections (V transposed)
    proj_kv_kernel<<<dim3(NTOT / 128, 2), 256, 0, stream>>>(
        x1, x2, WKb, bK, WVb, bV, Kb1, VbT1, Kb2, VbT2);
    // 4) scores + softmax + A + histogram
    scores_kernel<<<dim3(BGRAPHS * (MAXN / 128), 2), 256, 0, stream>>>(
        Kb1, Kb2, QpB, A1, A2, hist1, hist2);
    // 5) top-k thresholds
    select_kernel<<<1, 32, 0, stream>>>(hist1, hist2, thr);
    // 6) masked A@V + output projection + relu
    attn_out_kernel<<<dim3(BGRAPHS, 2), 256, 0, stream>>>(
        A1, A2, VbT1, VbT2, WOb, bO, thr, out);
}
